// GGNNEncoder_55508157333633
// MI455X (gfx1250) — matrix operand order repaired and checked by reading, not compiled, not run on hardware
//
#include <hip/hip_runtime.h>
#include <hip/hip_bf16.h>

typedef __attribute__((ext_vector_type(16))) __bf16 v16bf;
typedef __attribute__((ext_vector_type(8)))  float  v8f;

#define HDIM 128
#define NLAYERS 3

__device__ __forceinline__ unsigned short f2bf(float f) {
    unsigned int u = __float_as_uint(f);
    unsigned int r = (u + 0x7FFFu + ((u >> 16) & 1u)) >> 16;   // RNE
    return (unsigned short)r;
}

// ---------------------------------------------------------------------------
// Weight prep: fp32 -> bf16, transposed (Wt[n*K+k] = W[k*N+n])
// ---------------------------------------------------------------------------
__global__ __launch_bounds__(256)
void prep_transpose(const float* __restrict__ W, unsigned short* __restrict__ Wt,
                    int K, int N) {
    int i = blockIdx.x * 256 + threadIdx.x;
    if (i >= K * N) return;
    int n = i / K, k = i - n * K;
    Wt[i] = f2bf(W[k * N + n]);
}

__global__ __launch_bounds__(256)
void prep_copy(const float* __restrict__ W, unsigned short* __restrict__ Wt, int n) {
    int i = blockIdx.x * 256 + threadIdx.x;
    if (i < n) Wt[i] = f2bf(W[i]);
}

__global__ __launch_bounds__(256)
void zero_f32(float* __restrict__ p, int n) {
    int i = blockIdx.x * 256 + threadIdx.x;
    if (i < n) p[i] = 0.f;
}

// ---------------------------------------------------------------------------
// WMMA GEMM:  C[M, Hout] = act( A[M,128] @ B[128,Hout] + bias )
// Bt is B transposed, bf16: Bt[n*128 + k] = B[k][n]
// Block: 256 threads = 8 waves; block tile = 16 rows x 128 cols,
// each wave owns a 16x16 output tile; K=128 -> 4x wmma_f32_16x16x32_bf16.
// grid = (M/16, Hout/128)
// ---------------------------------------------------------------------------
__global__ __launch_bounds__(256, 2)
void wmma_gemm(const float* __restrict__ A, const unsigned short* __restrict__ Bt,
               const float* __restrict__ bias, float* __restrict__ C,
               int M, int Hout, int reluIn, int reluOut) {
    __shared__ alignas(16) unsigned short As[16 * HDIM];   // 4 KB bf16 A tile

    const int m0  = blockIdx.x * 16;
    const int n0  = blockIdx.y * 128;
    const int tid = threadIdx.x;

    // Stage A tile (16 x 128 fp32) into LDS as bf16 (optional input ReLU)
    for (int i = tid; i < 16 * HDIM; i += 256) {
        int r = i >> 7, c = i & 127;
        float v = A[(size_t)(m0 + r) * HDIM + c];
        if (reluIn) v = fmaxf(v, 0.f);
        As[i] = f2bf(v);
    }
    __syncthreads();

    const int wave = tid >> 5;
    const int lane = tid & 31;
    const int nw   = n0 + wave * 16;   // this wave's column-tile base
    const int mrow = lane & 15;        // A row / B column / C column
    const int hi   = lane >> 4;        // lane half

    v8f acc = {};
    #pragma unroll
    for (int kb = 0; kb < 4; ++kb) {
        // A fragment (ISA 16-bit A 16x32): lanes 0-15 K {0..7,16..23}, lanes 16-31 +8
        union { v16bf v; uint4 q[2]; } a;
        const int ka = kb * 32 + hi * 8;
        a.q[0] = *(const uint4*)&As[mrow * HDIM + ka];
        a.q[1] = *(const uint4*)&As[mrow * HDIM + ka + 16];
        // B fragment: column n = nw+mrow, K {0..7,16..23} (+8 for lanes 16-31), matching A
        union { v16bf v; uint4 q[2]; } b;
        const unsigned short* bp = Bt + (size_t)(nw + mrow) * HDIM + kb * 32 + hi * 8;
        b.q[0] = ((const uint4*)bp)[0];
        b.q[1] = ((const uint4*)bp)[2];
        acc = __builtin_amdgcn_wmma_f32_16x16x32_bf16(
                  false, a.v, false, b.v, (short)0, acc, false, false);
    }

    // Epilogue: lane holds rows hi*8 + r, column nw+mrow
    const int col = nw + mrow;
    float bv = bias ? bias[col] : 0.f;
    #pragma unroll
    for (int r = 0; r < 8; ++r) {
        float v = acc[r] + bv;
        if (reluOut) v = fmaxf(v, 0.f);
        C[(size_t)(m0 + hi * 8 + r) * Hout + col] = v;
    }
}

// ---------------------------------------------------------------------------
// Edge scatter-add: one wave per edge; lane handles 4 floats (float4 gather,
// 4x atomicAdd f32). t and agg are L2-resident (25.6 MB each, 192 MB L2).
// ---------------------------------------------------------------------------
__global__ __launch_bounds__(256)
void edge_scatter(const long long* __restrict__ ei, const float* __restrict__ t,
                  float* __restrict__ agg, int E) {
    int g = blockIdx.x * 256 + threadIdx.x;
    int e = g >> 5;
    if (e >= E) return;
    int lane = g & 31;
    long long s = ei[e];
    long long d = ei[(size_t)E + e];
    const float4 v = *(const float4*)&t[(size_t)s * HDIM + lane * 4];
    float* dp = &agg[(size_t)d * HDIM + lane * 4];
    atomicAdd(dp + 0, v.x);
    atomicAdd(dp + 1, v.y);
    atomicAdd(dp + 2, v.z);
    atomicAdd(dp + 3, v.w);
}

// ---------------------------------------------------------------------------
// GRU gate (torch.nn.GRUCell math), updates h in place.
// gi/gh are [N, 3H]; h is [N, H].
// ---------------------------------------------------------------------------
__global__ __launch_bounds__(256)
void gru_gate(const float* __restrict__ gi, const float* __restrict__ gh,
              float* __restrict__ h, int total) {
    int i = blockIdx.x * 256 + threadIdx.x;
    if (i >= total) return;
    int node = i >> 7, j = i & 127;
    size_t base = (size_t)node * (3 * HDIM);
    float r = 1.f / (1.f + __expf(-(gi[base + j]            + gh[base + j])));
    float z = 1.f / (1.f + __expf(-(gi[base + HDIM + j]     + gh[base + HDIM + j])));
    float n = tanhf(gi[base + 2 * HDIM + j] + r * gh[base + 2 * HDIM + j]);
    h[i] = (1.f - z) * n + z * h[i];
}

// ---------------------------------------------------------------------------
extern "C" void kernel_launch(void* const* d_in, const int* in_sizes, int n_in,
                              void* d_out, int out_size, void* d_ws, size_t ws_size,
                              hipStream_t stream) {
    const float*     x      = (const float*)d_in[0];
    const long long* ei     = (const long long*)d_in[1];
    const float*     W_in   = (const float*)d_in[2];
    const float*     b_in   = (const float*)d_in[3];
    const float*     ggnn_w = (const float*)d_in[4];
    const float*     w_ih   = (const float*)d_in[5];
    const float*     w_hh   = (const float*)d_in[6];
    const float*     b_ih   = (const float*)d_in[7];
    const float*     b_hh   = (const float*)d_in[8];
    const float*     W_out  = (const float*)d_in[9];
    const float*     b_out  = (const float*)d_in[10];
    float*           out    = (float*)d_out;

    const int N  = in_sizes[0] / HDIM;      // 50000 nodes total (B*n_nodes)
    const int E  = in_sizes[1] / 2;         // 600000 edges
    const int NF = N * HDIM;                // 6.4M floats

    // ---- workspace layout (bump allocator, 256B aligned) ----
    char* ws = (char*)d_ws;
    size_t off = 0;
    auto alloc = [&](size_t bytes) -> char* {
        char* p = ws + off;
        off += (bytes + 255) & ~(size_t)255;
        return p;
    };
    float* h   = (float*)alloc((size_t)NF * 4);
    float* t   = (float*)alloc((size_t)NF * 4);
    float* agg = (float*)alloc((size_t)NF * 4);
    float* gi  = (float*)alloc((size_t)N * 3 * HDIM * 4);
    float* gh  = (float*)alloc((size_t)N * 3 * HDIM * 4);
    unsigned short* Wt_in  = (unsigned short*)alloc(HDIM * HDIM * 2);
    unsigned short* Wt_g   = (unsigned short*)alloc((size_t)NLAYERS * HDIM * HDIM * 2);
    unsigned short* Wt_ih  = (unsigned short*)alloc(3 * HDIM * HDIM * 2);
    unsigned short* Wt_hh  = (unsigned short*)alloc(3 * HDIM * HDIM * 2);
    unsigned short* Wt_out = (unsigned short*)alloc(HDIM * HDIM * 2);
    (void)ws_size;

    const int WB = (HDIM * HDIM + 255) / 256;        // blocks for 128x128
    const int GB = (3 * HDIM * HDIM + 255) / 256;    // blocks for 384x128

    // ---- weight prep (bf16, B-transposed layout) ----
    prep_transpose<<<WB, 256, 0, stream>>>(W_in, Wt_in, HDIM, HDIM);
    for (int l = 0; l < NLAYERS; ++l)
        prep_transpose<<<WB, 256, 0, stream>>>(ggnn_w + (size_t)l * HDIM * HDIM,
                                               Wt_g + (size_t)l * HDIM * HDIM,
                                               HDIM, HDIM);
    prep_copy<<<GB, 256, 0, stream>>>(w_ih, Wt_ih, 3 * HDIM * HDIM);  // (w_ih.T)^T = w_ih
    prep_copy<<<GB, 256, 0, stream>>>(w_hh, Wt_hh, 3 * HDIM * HDIM);
    prep_transpose<<<WB, 256, 0, stream>>>(W_out, Wt_out, HDIM, HDIM);

    const dim3 gemm1(N / 16, 1);   // Hout = 128
    const dim3 gemm3(N / 16, 3);   // Hout = 384
    const int  eltB  = (NF + 255) / 256;
    const int  edgeB = (E * 32 + 255) / 256;

    // ---- fc_in: h = relu(x @ W_in + b_in) ----
    wmma_gemm<<<gemm1, 256, 0, stream>>>(x, Wt_in, b_in, h, N, HDIM, 0, 1);

    // ---- GGNN layers ----
    for (int l = 0; l < NLAYERS; ++l) {
        // t = h @ ggnn_w[l]
        wmma_gemm<<<gemm1, 256, 0, stream>>>(h, Wt_g + (size_t)l * HDIM * HDIM,
                                             nullptr, t, N, HDIM, 0, 0);
        // agg = segment_sum(t[src], dst)
        zero_f32<<<eltB, 256, 0, stream>>>(agg, NF);
        edge_scatter<<<edgeB, 256, 0, stream>>>(ei, t, agg, E);
        // gi = agg @ w_ih.T + b_ih ;  gh = h @ w_hh.T + b_hh
        wmma_gemm<<<gemm3, 256, 0, stream>>>(agg, Wt_ih, b_ih, gi, N, 3 * HDIM, 0, 0);
        wmma_gemm<<<gemm3, 256, 0, stream>>>(h,   Wt_hh, b_hh, gh, N, 3 * HDIM, 0, 0);
        // h = GRU(gi, gh, h)
        gru_gate<<<eltB, 256, 0, stream>>>(gi, gh, h, NF);
    }

    // ---- fc_out: out = relu(relu(h) @ W_out + b_out) ----
    wmma_gemm<<<gemm1, 256, 0, stream>>>(h, Wt_out, b_out, out, N, HDIM, 1, 1);
}
